// WindowAttention_33809982554257
// MI455X (gfx1250) — compile-verified, matrix operand order checked by reading, not run on hardware
//
#include <hip/hip_runtime.h>
#include <hip/hip_bf16.h>

typedef __attribute__((ext_vector_type(16))) _Float16 v16h;
typedef __attribute__((ext_vector_type(8)))  _Float16 v8h;
typedef __attribute__((ext_vector_type(8)))  float    v8f;

#define NTOK  49
#define NPAD  64
#define HEADS 16
#define HD    32
#define DIM   512
#define DIM3  1536
#define NWIN  64
#define QSCALE 0.17677669529663687f   // 32^-0.5

// Padded LDS pitches (bank-conflict-free, 16B-aligned rows):
#define XP  520      // x/Q/K/O row pitch in f16 (1040 B -> 4-bank row skew)
#define VTP 72       // sVT / sP row pitch in f16 (144 B -> 36-dword skew)

// LDS map (f16 units):
//   sX / sO : [64][520]          @ 0        (66560 B)
//   sQ      : [64][520] (scaled) @ 33280    (66560 B)
//   sK      : [64][520]          @ 66560    (66560 B)
//   sVT     : [16][32][72]       @ 99840    (73728 B)
//   sP      : 8 waves x [16][72] @ 136704   (18432 B)
// total = 291840 B  (<= 320 KB/WGP, one WG per WGP)
#define SMEM_BYTES 291840

__device__ __forceinline__ v16h mk16(v8h lo, v8h hi) {
    union { v16h v; v8h h[2]; } u;
    u.h[0] = lo; u.h[1] = hi;
    return u.v;
}
__device__ __forceinline__ v8h ld8(const _Float16* p) { return *(const v8h*)p; }

__global__ void cvt_weights_f16(const float* __restrict__ qkv_w,
                                const float* __restrict__ proj_w,
                                _Float16* __restrict__ w16)
{
    const int total = DIM3 * DIM + DIM * DIM;
    for (int i = blockIdx.x * blockDim.x + threadIdx.x; i < total;
         i += gridDim.x * blockDim.x) {
        float v = (i < DIM3 * DIM) ? qkv_w[i] : proj_w[i - DIM3 * DIM];
        w16[i] = (_Float16)v;
    }
}

__global__ void __launch_bounds__(256, 1)
win_attn_fused(const float* __restrict__ x,
               const float* __restrict__ mask,
               const float* __restrict__ rpb,
               const float* __restrict__ qkv_b,
               const float* __restrict__ proj_b,
               const _Float16* __restrict__ wq,   // (1536,512) f16
               const _Float16* __restrict__ wp,   // (512,512)  f16
               float* __restrict__ out)
{
    extern __shared__ _Float16 smem[];
    _Float16* sX  = smem;                 // reused as sO after attention
    _Float16* sQ  = smem + 33280;
    _Float16* sK  = smem + 66560;
    _Float16* sVT = smem + 99840;
    _Float16* sP  = smem + 136704;

    const int tid  = threadIdx.x;
    const int lane = tid & 31;
    const int wid  = tid >> 5;
    const int half = lane >> 4;           // A/B frag K-chunk select per ISA 7.12.2
    const int kh   = half * 8;
    const int l15  = lane & 15;
    const int b    = blockIdx.x;

    // ---------- Phase 0: x (49x512 f32) -> sX (64 rows x pitch 520, f16)
    {
        const float* xb = x + (size_t)b * NTOK * DIM;
        for (int v = tid; v < (NPAD * DIM) / 4; v += 256) {
            int row = v >> 7;                     // 128 float4 per logical row
            int c4  = (v & 127) << 2;
            float4 f = make_float4(0.f, 0.f, 0.f, 0.f);
            if (row < NTOK) f = *(const float4*)(xb + row * DIM + c4);
            _Float16* d = sX + row * XP + c4;
            d[0] = (_Float16)f.x; d[1] = (_Float16)f.y;
            d[2] = (_Float16)f.z; d[3] = (_Float16)f.w;
        }
    }
    __syncthreads();

    // ---------- Phase 1: QKV GEMM, register-blocked 4 M-tiles per N-tile
    // Each weight B-frag feeds 4 WMMAs (5 loads / 4 WMMAs).
    for (int nt = wid; nt < 96; nt += 8) {
        const int wrow = nt * 16 + l15;           // B col j = qkv_w row j
        v8f c[4] = {};
        #pragma unroll 2
        for (int ks = 0; ks < 16; ++ks) {
            const int k0 = ks * 32 + kh;
            v16h bf = mk16(ld8(wq + wrow * DIM + k0),
                           ld8(wq + wrow * DIM + k0 + 16));
            #pragma unroll
            for (int mt = 0; mt < 4; ++mt) {
                const int arow = mt * 16 + l15;
                v16h a = mk16(ld8(sX + arow * XP + k0),
                              ld8(sX + arow * XP + k0 + 16));
                c[mt] = __builtin_amdgcn_wmma_f32_16x16x32_f16(
                            false, a, false, bf, (short)0, c[mt], false, false);
            }
        }
        const int col   = nt * 16 + l15;
        const float bia = qkv_b[col];
        if (col < DIM) {                          // Q (pre-scaled)
            #pragma unroll
            for (int mt = 0; mt < 4; ++mt) {
                const int rbase = mt * 16 + half * 8;
                #pragma unroll
                for (int r = 0; r < 8; ++r)
                    sQ[(rbase + r) * XP + col] = (_Float16)((c[mt][r] + bia) * QSCALE);
            }
        } else if (col < 2 * DIM) {               // K
            #pragma unroll
            for (int mt = 0; mt < 4; ++mt) {
                const int rbase = mt * 16 + half * 8;
                #pragma unroll
                for (int r = 0; r < 8; ++r)
                    sK[(rbase + r) * XP + (col - DIM)] = (_Float16)(c[mt][r] + bia);
            }
        } else {                                  // V, transposed per head
            const int cv = col - 2 * DIM;
            const int h = cv >> 5, d = cv & 31;
            #pragma unroll
            for (int mt = 0; mt < 4; ++mt) {
                const int rbase = mt * 16 + half * 8;
                #pragma unroll
                for (int r = 0; r < 8; ++r)
                    sVT[h * (HD * VTP) + d * VTP + (rbase + r)] =
                        (_Float16)(c[mt][r] + bia);
            }
        }
    }
    __syncthreads();

    // ---------- Phase 2: attention, 2 heads per wave
    const float* maskw = mask + (size_t)(b & (NWIN - 1)) * NTOK * NTOK;
    _Float16* pW = sP + wid * (16 * VTP);

    for (int hi = 0; hi < 2; ++hi) {
        const int h = wid + hi * 8;
        const _Float16* qh = sQ + h * HD;
        const _Float16* kh2 = sK + h * HD;
        const _Float16* vt = sVT + h * (HD * VTP);

        for (int mt = 0; mt < 4; ++mt) {
            const int rbase = mt * 16 + half * 8;
            const int qrow  = mt * 16 + l15;
            // hoisted A-frag (invariant over nt)
            v16h aq = mk16(ld8(qh + qrow * XP + kh),
                           ld8(qh + qrow * XP + kh + 16));
            v8f acc[4];
            #pragma unroll
            for (int nt = 0; nt < 4; ++nt) {
                const int tcol = nt * 16 + l15;   // B col = token row of K
                v16h bf = mk16(ld8(kh2 + tcol * XP + kh),
                               ld8(kh2 + tcol * XP + kh + 16));
                v8f z = {};
                acc[nt] = __builtin_amdgcn_wmma_f32_16x16x32_f16(
                              false, aq, false, bf, (short)0, z, false, false);
            }
            // fused rel-pos bias + window mask + pad-column mask
            #pragma unroll
            for (int nt = 0; nt < 4; ++nt) {
                const int colg = nt * 16 + l15;
                #pragma unroll
                for (int r = 0; r < 8; ++r) {
                    const int rowg = rbase + r;
                    float s = acc[nt][r];
                    if (colg < NTOK && rowg < NTOK) {
                        const int r0 = rowg / 7, c0 = rowg % 7;
                        const int r1 = colg / 7, c1 = colg % 7;
                        const int idx = (r0 - r1 + 6) * 13 + (c0 - c1 + 6);
                        s += rpb[idx * HEADS + h] + maskw[rowg * NTOK + colg];
                    }
                    if (colg >= NTOK) s = -1e30f;
                    acc[nt][r] = s;
                }
            }
            // row softmax via width-16 segmented shuffles
            #pragma unroll
            for (int r = 0; r < 8; ++r) {
                float m = fmaxf(fmaxf(acc[0][r], acc[1][r]),
                                fmaxf(acc[2][r], acc[3][r]));
                for (int off = 1; off < 16; off <<= 1)
                    m = fmaxf(m, __shfl_xor(m, off, 16));
                float ssum = 0.f;
                #pragma unroll
                for (int nt = 0; nt < 4; ++nt) {
                    float e = __expf(acc[nt][r] - m);
                    acc[nt][r] = e;
                    ssum += e;
                }
                for (int off = 1; off < 16; off <<= 1)
                    ssum += __shfl_xor(ssum, off, 16);
                const float inv = 1.0f / ssum;
                #pragma unroll
                for (int nt = 0; nt < 4; ++nt)
                    pW[(half * 8 + r) * VTP + nt * 16 + l15] =
                        (_Float16)(acc[nt][r] * inv);
            }
            // O tile = P(16x64) @ V(64x32); hoisted P A-frags (invariant over n2)
            v16h ap[2];
            #pragma unroll
            for (int ks = 0; ks < 2; ++ks) {
                const int k0 = ks * 32 + kh;
                ap[ks] = mk16(ld8(pW + l15 * VTP + k0),
                              ld8(pW + l15 * VTP + k0 + 16));
            }
            #pragma unroll
            for (int n2 = 0; n2 < 2; ++n2) {
                const int d = n2 * 16 + l15;
                v8f o = {};
                #pragma unroll
                for (int ks = 0; ks < 2; ++ks) {
                    const int k0 = ks * 32 + kh;
                    v16h bf = mk16(ld8(vt + d * VTP + k0),
                                   ld8(vt + d * VTP + k0 + 16));
                    o = __builtin_amdgcn_wmma_f32_16x16x32_f16(
                            false, ap[ks], false, bf, (short)0, o, false, false);
                }
                #pragma unroll
                for (int r = 0; r < 8; ++r)
                    sX[(rbase + r) * XP + h * HD + n2 * 16 + l15] = (_Float16)(o[r]);
            }
        }
    }
    __syncthreads();

    // ---------- Phase 3: proj GEMM, register-blocked 4 M-tiles per N-tile
    float* ob = out + (size_t)b * NTOK * DIM;
    for (int nt = wid; nt < 32; nt += 8) {
        const int wrow = nt * 16 + l15;
        v8f c[4] = {};
        #pragma unroll 2
        for (int ks = 0; ks < 16; ++ks) {
            const int k0 = ks * 32 + kh;
            v16h bf = mk16(ld8(wp + wrow * DIM + k0),
                           ld8(wp + wrow * DIM + k0 + 16));
            #pragma unroll
            for (int mt = 0; mt < 4; ++mt) {
                const int arow = mt * 16 + l15;
                v16h a = mk16(ld8(sX + arow * XP + k0),
                              ld8(sX + arow * XP + k0 + 16));
                c[mt] = __builtin_amdgcn_wmma_f32_16x16x32_f16(
                            false, a, false, bf, (short)0, c[mt], false, false);
            }
        }
        const int col   = nt * 16 + l15;
        const float bia = proj_b[col];
        #pragma unroll
        for (int mt = 0; mt < 4; ++mt) {
            const int rbase = mt * 16 + half * 8;
            #pragma unroll
            for (int r = 0; r < 8; ++r) {
                const int row = rbase + r;
                if (row < NTOK) ob[row * DIM + col] = c[mt][r] + bia;
            }
        }
    }
}

extern "C" void kernel_launch(void* const* d_in, const int* in_sizes, int n_in,
                              void* d_out, int out_size, void* d_ws, size_t ws_size,
                              hipStream_t stream)
{
    const float* x      = (const float*)d_in[0];
    const float* mask   = (const float*)d_in[1];
    const float* rpb    = (const float*)d_in[2];
    const float* qkv_w  = (const float*)d_in[3];
    const float* qkv_b  = (const float*)d_in[4];
    const float* proj_w = (const float*)d_in[5];
    const float* proj_b = (const float*)d_in[6];
    float* out = (float*)d_out;

    _Float16* w16 = (_Float16*)d_ws;              // 2.1 MB of f16 weights
    const _Float16* wq = w16;
    const _Float16* wp = w16 + DIM3 * DIM;

    cvt_weights_f16<<<512, 256, 0, stream>>>(qkv_w, proj_w, w16);

    (void)hipFuncSetAttribute(reinterpret_cast<const void*>(win_attn_fused),
                              hipFuncAttributeMaxDynamicSharedMemorySize,
                              SMEM_BYTES);
    win_attn_fused<<<2048, 256, SMEM_BYTES, stream>>>(x, mask, rpb, qkv_b, proj_b,
                                                      wq, wp, out);
}
